// GatedDeltaNet_48928267436376
// MI455X (gfx1250) — matrix operand order repaired and checked by reading, not compiled, not run on hardware
//
#include <hip/hip_runtime.h>
#include <hip/hip_bf16.h>

// ---------------------------------------------------------------------------
// GatedDeltaNet forward for MI455X (gfx1250, wave32, WMMA bf16)
// ---------------------------------------------------------------------------

typedef __attribute__((ext_vector_type(16))) __bf16 v16bf;
typedef __attribute__((ext_vector_type(8)))  __bf16 v8bf;
typedef __attribute__((ext_vector_type(8)))  float  v8f;

#define BB   4
#define TT   2048
#define DD   2048
#define HH   16
#define DKH  128
#define MM   (BB * TT)          // 8192 rows

#define BM 128
#define BN 128
#define BK 32
#define LSTR 40                 // padded LDS row stride (elements)

// ---------------------------------------------------------------------------
// fp32 -> bf16 elementwise convert
// ---------------------------------------------------------------------------
__global__ __launch_bounds__(256)
void cvt_bf16_kernel(const float* __restrict__ src, __bf16* __restrict__ dst, int n) {
    int i = blockIdx.x * 256 + threadIdx.x;
    if (i < n) dst[i] = (__bf16)src[i];
}

// ---------------------------------------------------------------------------
// fp32 [R][C] -> bf16 [C][R] tiled transpose (weight -> Bt layout for GEMM)
// ---------------------------------------------------------------------------
__global__ __launch_bounds__(256)
void transpose_cvt_kernel(const float* __restrict__ src, __bf16* __restrict__ dst,
                          int R, int C) {
    __shared__ float tile[32][33];
    int rb = blockIdx.x * 32;
    int cb = blockIdx.y * 32;
    int tx = threadIdx.x, ty = threadIdx.y;   // block (32, 8)
#pragma unroll
    for (int j = 0; j < 4; ++j)
        tile[ty + j * 8][tx] = src[(size_t)(rb + ty + j * 8) * C + cb + tx];
    __syncthreads();
#pragma unroll
    for (int j = 0; j < 4; ++j)
        dst[(size_t)(cb + ty + j * 8) * R + rb + tx] = (__bf16)tile[tx][ty + j * 8];
}

// ---------------------------------------------------------------------------
// conv weight (O, C, K=4) fp32 -> bf16 Bt layout [O][j*2048 + c]
// ---------------------------------------------------------------------------
__global__ __launch_bounds__(256)
void conv_perm_kernel(const float* __restrict__ src, __bf16* __restrict__ dst, int n) {
    int idx = blockIdx.x * 256 + threadIdx.x;   // idx = o*8192 + j*2048 + c
    if (idx >= n) return;
    int o = idx >> 13;
    int rem = idx & 8191;
    int j = rem >> 11;
    int c = rem & 2047;
    dst[idx] = (__bf16)src[((size_t)o << 13) + (c << 2) + j];
}

// ---------------------------------------------------------------------------
// bf16 WMMA GEMM:  C[M][N] = A[M][Kd] * Bt[N][Kd]^T
//   epilogue: 0 = fp32 store, 1 = bf16 store, 2 = SiLU then fp32 store
//   convmode: A element (m, kk) = qp[m + (kk>>11) - 3][kk & 2047] (zero if t<0)
// ---------------------------------------------------------------------------
__global__ __launch_bounds__(256)
void gemm_bf16_wmma(const __bf16* __restrict__ A, const __bf16* __restrict__ Bt,
                    float* __restrict__ Cf, __bf16* __restrict__ Cb,
                    int M, int N, int Kd, int lda,
                    int epilogue, int convmode, int Tlen) {
    __shared__ __bf16 As[BM][LSTR];
    __shared__ __bf16 Bs[BN][LSTR];

    const int tid  = threadIdx.x;
    const int m0   = blockIdx.y * BM;
    const int n0   = blockIdx.x * BN;

    const int lrow = tid >> 1;            // 0..127  (tile row loaded by this thread)
    const int lcol = (tid & 1) * 16;      // 0 or 16 (16 bf16 = 32B per thread half)

    const int lane = tid & 31;
    const int wv   = tid >> 5;            // wave 0..7
    const int wm   = wv >> 1;             // 0..3  -> 32-row patch
    const int wn   = wv & 1;              // 0..1  -> 64-col patch
    const int lm   = lane & 15;
    const int lh   = lane >> 4;

    v8f acc[2][4];
#pragma unroll
    for (int mi = 0; mi < 2; ++mi)
#pragma unroll
        for (int ni = 0; ni < 4; ++ni)
#pragma unroll
            for (int j = 0; j < 8; ++j) acc[mi][ni][j] = 0.0f;

    const int arow = m0 + lrow;
    const int trow = arow & (Tlen - 1);   // Tlen is a power of two (2048)

    for (int k0 = 0; k0 < Kd; k0 += BK) {
        // ---- stage A tile (with causal time-shift in conv mode) ----
        {
            const __bf16* asrc;
            bool valid = true;
            if (convmode) {
                int shift = (k0 >> 11) - 3;                  // j - (K-1)
                valid = (trow + shift) >= 0;
                asrc = A + (size_t)(arow + shift) * lda + (k0 & 2047) + lcol;
            } else {
                asrc = A + (size_t)arow * lda + k0 + lcol;
            }
            uint4 d0, d1;
            if (valid) {
                d0 = ((const uint4*)asrc)[0];
                d1 = ((const uint4*)asrc)[1];
            } else {
                d0 = uint4{0u, 0u, 0u, 0u};
                d1 = uint4{0u, 0u, 0u, 0u};
            }
            *(uint4*)&As[lrow][lcol]     = d0;
            *(uint4*)&As[lrow][lcol + 8] = d1;

            const __bf16* bsrc = Bt + (size_t)(n0 + lrow) * Kd + k0 + lcol;
            uint4 b0 = ((const uint4*)bsrc)[0];
            uint4 b1 = ((const uint4*)bsrc)[1];
            *(uint4*)&Bs[lrow][lcol]     = b0;
            *(uint4*)&Bs[lrow][lcol + 8] = b1;
            if (k0 + BK < Kd) __builtin_prefetch((const char*)bsrc + BK * 2, 0, 1);
        }
        __syncthreads();

        // ---- gather fragments per documented gfx1250 WMMA layouts ----
        union V16 { v16bf v; v8bf h[2]; };
        V16 af[2], bfr[4];
#pragma unroll
        for (int mi = 0; mi < 2; ++mi) {
            int r = wm * 32 + mi * 16 + lm;
            af[mi].h[0] = *(const v8bf*)&As[r][lh * 8];        // K = lh*8 .. +7
            af[mi].h[1] = *(const v8bf*)&As[r][16 + lh * 8];   // K = 16+lh*8 .. +7
        }
#pragma unroll
        for (int ni = 0; ni < 4; ++ni) {
            int r = wn * 64 + ni * 16 + lm;
            bfr[ni].h[0] = *(const v8bf*)&Bs[r][lh * 8];       // K = lh*8 .. +7
            bfr[ni].h[1] = *(const v8bf*)&Bs[r][16 + lh * 8];  // K = 16+lh*8 .. +7
        }
#pragma unroll
        for (int mi = 0; mi < 2; ++mi)
#pragma unroll
            for (int ni = 0; ni < 4; ++ni)
                acc[mi][ni] = __builtin_amdgcn_wmma_f32_16x16x32_bf16(
                    false, af[mi].v, false, bfr[ni].v,
                    (short)0, acc[mi][ni], false, false);
        __syncthreads();
    }

    // ---- epilogue (C/D layout: row = j + 8*lh, col = lane%16) ----
#pragma unroll
    for (int mi = 0; mi < 2; ++mi) {
#pragma unroll
        for (int ni = 0; ni < 4; ++ni) {
            int grb = m0 + wm * 32 + mi * 16 + lh * 8;
            int gc  = n0 + wn * 64 + ni * 16 + lm;
#pragma unroll
            for (int j = 0; j < 8; ++j) {
                float v = acc[mi][ni][j];
                if (epilogue == 2) v = v / (1.0f + __expf(-v));     // SiLU
                size_t idx = (size_t)(grb + j) * N + gc;
                if (epilogue == 1) Cb[idx] = (__bf16)v;
                else               Cf[idx] = v;
            }
        }
    }
}

// ---------------------------------------------------------------------------
// beta = sigmoid(x @ Wbeta); alpha = exp(-exp(logA) * softplus(x @ Walpha + dt))
// stored (B*T, H) fp32
// ---------------------------------------------------------------------------
__global__ __launch_bounds__(256)
void beta_alpha_kernel(const float* __restrict__ x,
                       const float* __restrict__ Wb, const float* __restrict__ Wa,
                       const float* __restrict__ logA, const float* __restrict__ dt,
                       float* __restrict__ beta, float* __restrict__ alpha) {
    int r = threadIdx.x >> 4;          // 0..15 row-in-block
    int h = threadIdx.x & 15;          // head
    int m = blockIdx.x * 16 + r;
    const float* xr = x + (size_t)m * DD;
    float sb = 0.0f, sa = 0.0f;
    for (int k = 0; k < DD; ++k) {
        float xv = xr[k];
        sb = fmaf(xv, Wb[k * HH + h], sb);
        sa = fmaf(xv, Wa[k * HH + h], sa);
    }
    float bt = 1.0f / (1.0f + __expf(-sb));
    float z  = sa + dt[h];
    float sp = (z > 0.0f) ? (z + log1pf(__expf(-z))) : log1pf(__expf(z));
    float al = __expf(-__expf(logA[h]) * sp);
    beta[(size_t)m * HH + h]  = bt;
    alpha[(size_t)m * HH + h] = al;
}

// ---------------------------------------------------------------------------
// per-(b,t,head) L2 norm over 128 channels, in place, times outscale
// block: 512 threads = 16 waves; wave w handles head w; 4 elems / lane
// ---------------------------------------------------------------------------
__global__ __launch_bounds__(512)
void l2norm_kernel(float* __restrict__ buf, float outscale) {
    int m    = blockIdx.x;
    int head = threadIdx.x >> 5;
    int lane = threadIdx.x & 31;
    float* p = buf + (size_t)m * DD + head * DKH + lane * 4;
    float4 v = *(const float4*)p;
    float s = v.x * v.x + v.y * v.y + v.z * v.z + v.w * v.w;
#pragma unroll
    for (int off = 16; off >= 1; off >>= 1) s += __shfl_xor(s, off, 32);
    float inv = outscale / fmaxf(sqrtf(s), 1e-6f);
    v.x *= inv; v.y *= inv; v.z *= inv; v.w *= inv;
    *(float4*)p = v;
}

// ---------------------------------------------------------------------------
// Gated delta rule, sequential over T. One block per (b,h), 128 threads.
// S[v][k]: thread v holds row v (128 fp32 registers). ctx overwrites v in place.
// ---------------------------------------------------------------------------
__global__ __launch_bounds__(128)
void delta_rule_kernel(const float* __restrict__ qb, const float* __restrict__ kb,
                       float* __restrict__ vb,
                       const float* __restrict__ beta, const float* __restrict__ alpha) {
    const int b   = blockIdx.x >> 4;
    const int h   = blockIdx.x & 15;
    const int tid = threadIdx.x;       // v index 0..127
    __shared__ float ksh[DKH];
    __shared__ float qsh[DKH];

    float S[DKH];
#pragma unroll
    for (int i = 0; i < DKH; ++i) S[i] = 0.0f;

    const size_t base = ((size_t)b * TT) * DD + (size_t)h * DKH;
    for (int t = 0; t < TT; ++t) {
        const size_t off = base + (size_t)t * DD;
        __syncthreads();                       // protect previous-step reads
        ksh[tid] = kb[off + tid];
        qsh[tid] = qb[off + tid];
        float vt = vb[off + tid];
        float a  = alpha[((size_t)b * TT + t) * HH + h];
        float bt = beta [((size_t)b * TT + t) * HH + h];
        __syncthreads();

        float dotk = 0.0f;
#pragma unroll
        for (int k = 0; k < DKH; ++k) dotk = fmaf(S[k], ksh[k], dotk);
        float delta = bt * (vt - a * dotk);

        float o = 0.0f;
#pragma unroll
        for (int k = 0; k < DKH; ++k) {
            float sn = fmaf(a, S[k], delta * ksh[k]);   // S = a*S + delta (x) k
            S[k] = sn;
            o = fmaf(sn, qsh[k], o);
        }
        vb[off + tid] = o;                     // ctx in place
    }
}

// ---------------------------------------------------------------------------
// ZeroCenteredRMSNorm * (1+scale) * SiLU(gate) -> bf16 (input of output GEMM)
// block 256, 8 elems / thread, one (b,t) row per block
// ---------------------------------------------------------------------------
__global__ __launch_bounds__(256)
void rmsgate_kernel(const float* __restrict__ ctx, const float* __restrict__ gate,
                    const float* __restrict__ scale, __bf16* __restrict__ outb) {
    int m   = blockIdx.x;
    int tid = threadIdx.x;
    __shared__ float sh[8];
    size_t base = (size_t)m * DD + tid * 8;
    float4 p0 = *(const float4*)(ctx + base);
    float4 p1 = *(const float4*)(ctx + base + 4);
    float v[8] = {p0.x, p0.y, p0.z, p0.w, p1.x, p1.y, p1.z, p1.w};
    float s = 0.0f;
#pragma unroll
    for (int i = 0; i < 8; ++i) s = fmaf(v[i], v[i], s);
#pragma unroll
    for (int off = 16; off >= 1; off >>= 1) s += __shfl_xor(s, off, 32);
    if ((tid & 31) == 0) sh[tid >> 5] = s;
    __syncthreads();
    float tot = 0.0f;
#pragma unroll
    for (int i = 0; i < 8; ++i) tot += sh[i];
    float rms = rsqrtf(tot / (float)DD + 1e-6f);
#pragma unroll
    for (int i = 0; i < 8; ++i) {
        int c = tid * 8 + i;
        float g = gate[(size_t)m * DD + c];
        g = g / (1.0f + __expf(-g));
        outb[(size_t)m * DD + c] = (__bf16)(v[i] * rms * (1.0f + scale[c]) * g);
    }
}

// ---------------------------------------------------------------------------
// host-side orchestration
// ---------------------------------------------------------------------------
extern "C" void kernel_launch(void* const* d_in, const int* in_sizes, int n_in,
                              void* d_out, int out_size, void* d_ws, size_t ws_size,
                              hipStream_t stream) {
    (void)in_sizes; (void)n_in; (void)out_size; (void)ws_size;
    const float* x      = (const float*)d_in[0];
    const float* Wq     = (const float*)d_in[1];
    const float* Wk     = (const float*)d_in[2];
    const float* Wv     = (const float*)d_in[3];
    const float* Wbeta  = (const float*)d_in[4];
    const float* Walpha = (const float*)d_in[5];
    const float* convq  = (const float*)d_in[6];
    const float* convk  = (const float*)d_in[7];
    const float* convv  = (const float*)d_in[8];
    const float* logA   = (const float*)d_in[9];
    const float* dt     = (const float*)d_in[10];
    const float* scale  = (const float*)d_in[11];
    const float* Wgate  = (const float*)d_in[12];
    const float* Wout   = (const float*)d_in[13];
    float* out = (float*)d_out;

    const size_t SZ_XB = (size_t)MM * DD * 2;       // x bf16
    const size_t SZ_WT = (size_t)DD * DD * 2;       // transposed weight bf16
    const size_t SZ_CT = (size_t)DD * 4 * DD * 2;   // conv Bt bf16 (O x 8192)
    const size_t SZ_PB = (size_t)MM * DD * 2;       // projection bf16
    const size_t SZ_CF = (size_t)MM * DD * 4;       // fp32 activation
    const size_t SZ_BA = (size_t)MM * HH * 4;       // beta/alpha

    unsigned char* ws = (unsigned char*)d_ws;
    size_t off = 0;
    auto take = [&](size_t b) { unsigned char* p = ws + off;
                                off += (b + 255) & ~(size_t)255; return p; };
    __bf16* xb  = (__bf16*)take(SZ_XB);
    __bf16* WqT = (__bf16*)take(SZ_WT);
    __bf16* WkT = (__bf16*)take(SZ_WT);
    __bf16* WvT = (__bf16*)take(SZ_WT);
    __bf16* WgT = (__bf16*)take(SZ_WT);
    __bf16* WoT = (__bf16*)take(SZ_WT);
    __bf16* CqT = (__bf16*)take(SZ_CT);
    __bf16* CkT = (__bf16*)take(SZ_CT);
    __bf16* CvT = (__bf16*)take(SZ_CT);
    __bf16* qp  = (__bf16*)take(SZ_PB);     // later reused for gated bf16
    __bf16* kp  = (__bf16*)take(SZ_PB);
    __bf16* vp  = (__bf16*)take(SZ_PB);
    float*  qc  = (float*)take(SZ_CF);
    float*  kc  = (float*)take(SZ_CF);      // later reused for gate fp32
    float*  vc  = (float*)take(SZ_CF);      // becomes ctx in place
    float*  betab  = (float*)take(SZ_BA);
    float*  alphab = (float*)take(SZ_BA);

    // ---- 0. conversions ----
    {
        int n = MM * DD;
        cvt_bf16_kernel<<<dim3(n / 256), dim3(256), 0, stream>>>(x, xb, n);
        dim3 tb(32, 8), tg(DD / 32, DD / 32);
        transpose_cvt_kernel<<<tg, tb, 0, stream>>>(Wq,   WqT, DD, DD);
        transpose_cvt_kernel<<<tg, tb, 0, stream>>>(Wk,   WkT, DD, DD);
        transpose_cvt_kernel<<<tg, tb, 0, stream>>>(Wv,   WvT, DD, DD);
        transpose_cvt_kernel<<<tg, tb, 0, stream>>>(Wgate, WgT, DD, DD);
        transpose_cvt_kernel<<<tg, tb, 0, stream>>>(Wout, WoT, DD, DD);
        int nc = DD * DD * 4;
        conv_perm_kernel<<<dim3(nc / 256), dim3(256), 0, stream>>>(convq, CqT, nc);
        conv_perm_kernel<<<dim3(nc / 256), dim3(256), 0, stream>>>(convk, CkT, nc);
        conv_perm_kernel<<<dim3(nc / 256), dim3(256), 0, stream>>>(convv, CvT, nc);
    }

    dim3 gg(DD / BN, MM / BM);   // (16, 64)

    // ---- 1. q/k/v projections (bf16 out) ----
    gemm_bf16_wmma<<<gg, 256, 0, stream>>>(xb, WqT, nullptr, qp, MM, DD, DD, DD, 1, 0, TT);
    gemm_bf16_wmma<<<gg, 256, 0, stream>>>(xb, WkT, nullptr, kp, MM, DD, DD, DD, 1, 0, TT);
    gemm_bf16_wmma<<<gg, 256, 0, stream>>>(xb, WvT, nullptr, vp, MM, DD, DD, DD, 1, 0, TT);

    // ---- 2. causal dense convs as shifted GEMMs + SiLU (fp32 out) ----
    gemm_bf16_wmma<<<gg, 256, 0, stream>>>(qp, CqT, qc, nullptr, MM, DD, 4 * DD, DD, 2, 1, TT);
    gemm_bf16_wmma<<<gg, 256, 0, stream>>>(kp, CkT, kc, nullptr, MM, DD, 4 * DD, DD, 2, 1, TT);
    gemm_bf16_wmma<<<gg, 256, 0, stream>>>(vp, CvT, vc, nullptr, MM, DD, 4 * DD, DD, 2, 1, TT);

    // ---- 3. beta / alpha ----
    beta_alpha_kernel<<<dim3(MM / 16), dim3(256), 0, stream>>>(x, Wbeta, Walpha,
                                                               logA, dt, betab, alphab);

    // ---- 4. per-head L2 norm (q also gets dk^-0.5) ----
    l2norm_kernel<<<dim3(MM), dim3(512), 0, stream>>>(qc, 0.08838834764831845f); // 1/sqrt(128)
    l2norm_kernel<<<dim3(MM), dim3(512), 0, stream>>>(kc, 1.0f);

    // ---- 5. sequential gated delta rule (ctx overwrites vc) ----
    delta_rule_kernel<<<dim3(BB * HH), dim3(128), 0, stream>>>(qc, kc, vc, betab, alphab);

    // ---- 6. gate = SiLU(x @ Wgate)  (reuses kc after scan consumed it) ----
    gemm_bf16_wmma<<<gg, 256, 0, stream>>>(xb, WgT, kc, nullptr, MM, DD, DD, DD, 2, 0, TT);

    // ---- 7. RMSNorm * (1+scale) * gate -> bf16 (reuses qp) ----
    rmsgate_kernel<<<dim3(MM), dim3(256), 0, stream>>>(vc, kc, scale, qp);

    // ---- 8. output projection -> d_out fp32 ----
    gemm_bf16_wmma<<<gg, 256, 0, stream>>>(qp, WoT, out, nullptr, MM, DD, DD, DD, 0, 0, TT);
}
